// DPA3NextLayer_22402549416332
// MI455X (gfx1250) — matrix operand order repaired and checked by reading, not compiled, not run on hardware
//
#include <hip/hip_runtime.h>
#include <hip/hip_bf16.h>
#include <math.h>

// ---------------------------------------------------------------- types
typedef __bf16 bf16_t;
typedef bf16_t v16bf __attribute__((ext_vector_type(16)));
typedef float  v8f   __attribute__((ext_vector_type(8)));
typedef unsigned int u32x4 __attribute__((ext_vector_type(4)));

#define DEVFN __device__ __forceinline__

// ---------------------------------------------------------------- problem sizes
static constexpr int kNF = 2, kNLOC = 4096, kNALL = 6144;
static constexpr int kNEDGE = 131072, kNANGLE = 262144;
static constexpr int kNNODE = kNF * kNLOC;   // 8192
static constexpr int kNEXT  = kNF * kNALL;   // 12288
static constexpr float kASCALE = 0.5f;                  // (40/10)^-0.5
static constexpr float kESCALE = 0.2886751345948129f;   // (120/10)^-0.5

// ---------------------------------------------------------------- ws layout (bytes)
static constexpr size_t WT_LA_IN   = 0;        // 448x512
static constexpr size_t WT_LA_OUT  = 458752;   // 256x128
static constexpr size_t WT_LA_GATE = 524288;   // 448x128
static constexpr size_t WT_AA_IN   = 638976;   // 384x256
static constexpr size_t WT_AA_OUT  = 835584;   // 128x128
static constexpr size_t WT_AA_SRCG = 868352;   // 128x128
static constexpr size_t WT_AE_IN   = 901120;   // 384x512
static constexpr size_t WT_AE_OUT  = 1294336;  // 256x128
static constexpr size_t WT_LR_IN   = 1359872;  // 448x512
static constexpr size_t WT_LR_OUT  = 1818624;  // 256x128
static constexpr size_t WT_LR_EPROJ= 1884160;  // 128x128
static constexpr size_t WT_LR_APROJ= 1916928;  // 128x64
static constexpr size_t WT_AR_IN   = 1933312;  // 384x256
static constexpr size_t WT_AR_OUT  = 2129920;  // 128x128
static constexpr size_t WT_AR_NPROJ= 2162688;  // 128x128
static constexpr size_t WT_AR_EPROJ= 2195456;  // 128x128

static constexpr size_t O_EDGE_NRM   = 4194304;                    // bf16 NEDGE*128
static constexpr size_t O_ANGLE_NRM  = O_EDGE_NRM  + 33554432;     // bf16 NANGLE*64
static constexpr size_t O_NODE_BF    = O_ANGLE_NRM + 33554432;     // bf16 NNODE*128
static constexpr size_t O_NODEEXT_BF = O_NODE_BF   + 2097152;      // bf16 NEXT*128
static constexpr size_t O_MSG_BF     = O_NODEEXT_BF+ 3145728;      // bf16 NANGLE*128
static constexpr size_t O_EUPD_BF    = O_MSG_BF    + 67108864;     // bf16 NEDGE*128
static constexpr size_t O_F          = O_EUPD_BF   + 33554432;     // f32  NANGLE*128 (also ENV)
static constexpr size_t O_SEGM       = O_F         + 134217728;    // f32  NEDGE*128 (max / deltas)
static constexpr size_t O_SEGS       = O_SEGM      + 67108864;     // f32  NEDGE*128 (sum)
static constexpr size_t WS_NEED      = O_SEGS      + 67108864;     // ~425 MiB

// ---------------------------------------------------------------- scalar helpers
DEVFN unsigned short f2bf(float f) {
  unsigned int u = __float_as_uint(f);
  unsigned int r = u + 0x7FFFu + ((u >> 16) & 1u);
  return (unsigned short)(r >> 16);
}
DEVFN float bf2f(unsigned short h) { return __uint_as_float(((unsigned int)h) << 16); }
DEVFN float sigm(float x) { return 1.f / (1.f + expf(-x)); }

DEVFN void atomicMaxFloat(float* addr, float val) {
  if (val >= 0.f) atomicMax((int*)addr, __float_as_int(val));
  else            atomicMin((unsigned int*)addr, __float_as_uint(val));
}

// ---------------------------------------------------------------- WMMA helpers
DEVFN v8f wmma_bf16(v16bf a, v16bf b, v8f c) {
  return __builtin_amdgcn_wmma_f32_16x16x32_bf16(false, a, false, b, (short)0, c, false, false);
}

union V16U { v16bf v; u32x4 u[2]; };

// A tile (16x32, 16-bit): lane holds row M=lane&15; lanes0-15 K={0..7,16..23}, lanes16-31 K={8..15,24..31}
DEVFN v16bf load_tileA(const unsigned short* base, int pitch, int kstep, int lane) {
  int row = lane & 15;
  int kb  = kstep * 32 + ((lane & 16) ? 8 : 0);
  const u32x4* p = (const u32x4*)(base + (size_t)row * pitch + kb);
  V16U t; t.u[0] = p[0]; t.u[1] = p[2]; // +16 halves
  return t.v;
}
// B tile (32x16, 16-bit) from W^T (N x K row-major): lane holds col N=lane&15; lanes0-15 K={0..7,16..23}, lanes16-31 K={8..15,24..31}
DEVFN v16bf load_tileB16(const unsigned short* wT, int pitch, int n0, int kstep, int lane) {
  int n  = n0 + (lane & 15);
  int kb = kstep * 32 + ((lane & 16) ? 8 : 0);
  const u32x4* p = (const u32x4*)(wT + (size_t)n * pitch + kb);
  V16U t; t.u[0] = p[0]; t.u[1] = p[2];
  return t.v;
}

template <int K, int PITCH>
DEVFN v8f tile_matmul_out128(const unsigned short* sX, const unsigned short* wT, int wave, int lane) {
  v8f acc = {0.f, 0.f, 0.f, 0.f, 0.f, 0.f, 0.f, 0.f};
#pragma unroll
  for (int ks = 0; ks < K / 32; ks++) {
    v16bf a = load_tileA(sX, PITCH, ks, lane);
    v16bf b = load_tileB16(wT, K, wave * 16, ks, lane);
    acc = wmma_bf16(a, b, acc);
  }
  return acc;
}

// ---------------------------------------------------------------- gathers (16 rows into LDS bf16)
DEVFN void gather_angle(unsigned short* sA, int a0,
                        const unsigned short* __restrict__ ang_nrm,
                        const unsigned short* __restrict__ node_bf,
                        const unsigned short* __restrict__ edge_nrm,
                        const int* __restrict__ n2a, const int* __restrict__ eij,
                        const int* __restrict__ eik) {
  for (int c = threadIdx.x; c < 16 * 56; c += 256) {
    int row = c / 56, seg = c % 56;
    int a = a0 + row;
    const unsigned short* src; int off;
    if (seg < 8)       { src = ang_nrm  + (size_t)a * 64;         off = seg * 8; }
    else if (seg < 24) { src = node_bf  + (size_t)n2a[a] * 128;   off = (seg - 8) * 8; }
    else if (seg < 40) { src = edge_nrm + (size_t)eij[a] * 128;   off = (seg - 24) * 8; }
    else               { src = edge_nrm + (size_t)eik[a] * 128;   off = (seg - 40) * 8; }
    *(u32x4*)(sA + row * 448 + seg * 8) = *(const u32x4*)(src + off);
  }
}

DEVFN void gather_edge(unsigned short* sA, int e0,
                       const unsigned short* __restrict__ edge_nrm,
                       const unsigned short* __restrict__ node_nrm,
                       const unsigned short* __restrict__ nodeext_bf,
                       const int* __restrict__ n2e, const int* __restrict__ n_ext2e) {
  for (int c = threadIdx.x; c < 16 * 48; c += 256) {
    int row = c / 48, seg = c % 48;
    int e = e0 + row;
    const unsigned short* src; int off;
    if (seg < 16)      { src = edge_nrm   + (size_t)e * 128;           off = seg * 8; }
    else if (seg < 32) { src = node_nrm   + (size_t)n2e[e] * 128;      off = (seg - 16) * 8; }
    else               { src = nodeext_bf + (size_t)n_ext2e[e] * 128;  off = (seg - 32) * 8; }
    *(u32x4*)(sA + row * 384 + seg * 8) = *(const u32x4*)(src + off);
  }
}

// ---------------------------------------------------------------- gated MLP core (h = A@Win, g = silu-gate, rmsnorm -> sG bf16)
template <int KIN, int H>
DEVFN void gated_mlp_core(const unsigned short* sA, float* sH, unsigned short* sG,
                          const unsigned short* __restrict__ w_inT,
                          const float* __restrict__ normw, int wave, int lane) {
  constexpr int H2 = 2 * H;
  constexpr int NT = H2 / 128;   // n-tiles per wave (8 waves cover H2 cols)
  constexpr int KS = KIN / 32;
  v8f acc[NT];
#pragma unroll
  for (int i = 0; i < NT; i++) acc[i] = (v8f){0.f, 0.f, 0.f, 0.f, 0.f, 0.f, 0.f, 0.f};
  int n0w = wave * (H2 / 8);
  for (int ks = 0; ks < KS; ks++) {
    v16bf a = load_tileA(sA, KIN, ks, lane);
#pragma unroll
    for (int nt = 0; nt < NT; nt++) {
      v16bf b = load_tileB16(w_inT, KIN, n0w + nt * 16, ks, lane);
      acc[nt] = wmma_bf16(a, b, acc[nt]);
    }
  }
  int rofs = (lane >> 4) << 3;
#pragma unroll
  for (int nt = 0; nt < NT; nt++)
#pragma unroll
    for (int r = 0; r < 8; r++)
      sH[(r + rofs) * H2 + n0w + nt * 16 + (lane & 15)] = acc[nt][r];
  __syncthreads();

  constexpr int PER = H / 32;
  for (int rr = 0; rr < 2; rr++) {
    int row = wave * 2 + rr;
    float g[PER]; float ss = 0.f;
#pragma unroll
    for (int j = 0; j < PER; j++) {
      int c = lane + j * 32;
      float v  = sH[row * H2 + c];
      float gt = sH[row * H2 + H + c];
      float x  = v * gt / (1.f + expf(-gt));   // value * silu(gate)
      g[j] = x; ss += x * x;
    }
#pragma unroll
    for (int off = 16; off; off >>= 1) ss += __shfl_xor(ss, off, 32);
    float inv = rsqrtf(ss / (float)H + 1e-6f);
#pragma unroll
    for (int j = 0; j < PER; j++) {
      int c = lane + j * 32;
      sG[row * H + c] = f2bf(g[j] * inv * normw[c]);
    }
  }
  __syncthreads();
}

// ---------------------------------------------------------------- big fused kernels
__global__ __launch_bounds__(256) void k_line_attn(
    const unsigned short* __restrict__ ang_nrm, const unsigned short* __restrict__ node_bf,
    const unsigned short* __restrict__ edge_nrm,
    const int* __restrict__ n2a, const int* __restrict__ eij, const int* __restrict__ eik,
    const unsigned short* __restrict__ w_inT, const float* __restrict__ normw,
    const unsigned short* __restrict__ w_outT, const unsigned short* __restrict__ gateT,
    unsigned short* __restrict__ msg_bf, float* __restrict__ fbuf, float* __restrict__ segmax) {
  __shared__ alignas(16) unsigned short sA[16 * 448];
  __shared__ alignas(16) float          sH[16 * 512];
  __shared__ alignas(16) unsigned short sG[16 * 256];
  int wave = threadIdx.x >> 5, lane = threadIdx.x & 31;
  int a0 = blockIdx.x * 16;
  gather_angle(sA, a0, ang_nrm, node_bf, edge_nrm, n2a, eij, eik);
  __syncthreads();
  gated_mlp_core<448, 256>(sA, sH, sG, w_inT, normw, wave, lane);
  v8f msg = tile_matmul_out128<256, 256>(sG, w_outT, wave, lane);
  v8f f   = tile_matmul_out128<448, 448>(sA, gateT, wave, lane);
  int col = wave * 16 + (lane & 15);
  int rofs = (lane >> 4) << 3;
#pragma unroll
  for (int r = 0; r < 8; r++) {
    int a = a0 + rofs + r;
    msg_bf[(size_t)a * 128 + col] = f2bf(msg[r]);
    float fv = f[r];
    fbuf[(size_t)a * 128 + col] = fv;
    atomicMaxFloat(&segmax[(size_t)eij[a] * 128 + col], fv);
  }
}

__global__ __launch_bounds__(256) void k_line_ref(
    const unsigned short* __restrict__ ang_nrm, const unsigned short* __restrict__ node_bf,
    const unsigned short* __restrict__ edge_nrm,
    const int* __restrict__ n2a, const int* __restrict__ eij, const int* __restrict__ eik,
    const unsigned short* __restrict__ w_inT, const float* __restrict__ normw,
    const unsigned short* __restrict__ w_outT,
    const float* __restrict__ env, const float* __restrict__ a_sw,
    unsigned short* __restrict__ upd_bf, float* __restrict__ delta_e) {
  __shared__ alignas(16) unsigned short sA[16 * 448];
  __shared__ alignas(16) float          sH[16 * 512];
  __shared__ alignas(16) unsigned short sG[16 * 256];
  int wave = threadIdx.x >> 5, lane = threadIdx.x & 31;
  int a0 = blockIdx.x * 16;
  gather_angle(sA, a0, ang_nrm, node_bf, edge_nrm, n2a, eij, eik);
  __syncthreads();
  gated_mlp_core<448, 256>(sA, sH, sG, w_inT, normw, wave, lane);
  v8f upd = tile_matmul_out128<256, 256>(sG, w_outT, wave, lane);
  int col = wave * 16 + (lane & 15);
  int rofs = (lane >> 4) << 3;
#pragma unroll
  for (int r = 0; r < 8; r++) {
    int a  = a0 + rofs + r;
    int te = eij[a];
    float u = upd[r];
    upd_bf[(size_t)a * 128 + col] = f2bf(u);
    float ee = sigm(env[(size_t)te * 128 + col]) * sigm(env[(size_t)eik[a] * 128 + col]);
    atomicAdd(&delta_e[(size_t)te * 128 + col], u * ee * a_sw[a]);
  }
}

__global__ __launch_bounds__(256) void k_atom_attn(
    const unsigned short* __restrict__ edge_nrm, const unsigned short* __restrict__ node_nrm,
    const unsigned short* __restrict__ nodeext_bf,
    const int* __restrict__ n2e, const int* __restrict__ n_ext2e,
    const unsigned short* __restrict__ w1_inT, const float* __restrict__ norm1,
    const unsigned short* __restrict__ w1_outT, const unsigned short* __restrict__ srcgT,
    const unsigned short* __restrict__ w2_inT, const float* __restrict__ norm2,
    const unsigned short* __restrict__ w2_outT,
    unsigned short* __restrict__ msg_bf, float* __restrict__ fbuf, float* __restrict__ nodemax,
    const float* __restrict__ sw, float* __restrict__ edge_out) {
  __shared__ alignas(16) unsigned short sA[16 * 384];
  __shared__ alignas(16) float          sH[16 * 512];
  __shared__ alignas(16) unsigned short sG[16 * 256];
  int wave = threadIdx.x >> 5, lane = threadIdx.x & 31;
  int e0 = blockIdx.x * 16;
  gather_edge(sA, e0, edge_nrm, node_nrm, nodeext_bf, n2e, n_ext2e);
  __syncthreads();
  // src-gate logits from the edge_normed slice of the gathered features
  v8f g = tile_matmul_out128<128, 384>(sA, srcgT, wave, lane);
  // MLP1: node message
  gated_mlp_core<384, 128>(sA, sH, sG, w1_inT, norm1, wave, lane);
  v8f m1 = tile_matmul_out128<128, 128>(sG, w1_outT, wave, lane);
  int col = wave * 16 + (lane & 15);
  int rofs = (lane >> 4) << 3;
#pragma unroll
  for (int r = 0; r < 8; r++) {
    int e = e0 + rofs + r;
    msg_bf[(size_t)e * 128 + col] = f2bf(m1[r]);
    float gv = g[r];
    fbuf[(size_t)e * 128 + col] = gv;
    atomicMaxFloat(&nodemax[(size_t)n2e[e] * 128 + col], gv);
  }
  __syncthreads();
  // MLP2: edge self-update
  gated_mlp_core<384, 256>(sA, sH, sG, w2_inT, norm2, wave, lane);
  v8f m2 = tile_matmul_out128<256, 256>(sG, w2_outT, wave, lane);
#pragma unroll
  for (int r = 0; r < 8; r++) {
    int e = e0 + rofs + r;
    size_t o = (size_t)e * 128 + col;
    edge_out[o] += m2[r] * sw[e];
  }
}

__global__ __launch_bounds__(256) void k_atom_ref(
    const unsigned short* __restrict__ edge_nrm, const unsigned short* __restrict__ node_nrm,
    const unsigned short* __restrict__ nodeext_bf,
    const int* __restrict__ n2e, const int* __restrict__ n_ext2e,
    const unsigned short* __restrict__ w_inT, const float* __restrict__ normw,
    const unsigned short* __restrict__ w_outT,
    const float* __restrict__ sw, const float* __restrict__ env,
    unsigned short* __restrict__ upd_bf, float* __restrict__ node_delta) {
  __shared__ alignas(16) unsigned short sA[16 * 384];
  __shared__ alignas(16) float          sH[16 * 512];
  __shared__ alignas(16) unsigned short sG[16 * 256];
  int wave = threadIdx.x >> 5, lane = threadIdx.x & 31;
  int e0 = blockIdx.x * 16;
  gather_edge(sA, e0, edge_nrm, node_nrm, nodeext_bf, n2e, n_ext2e);
  __syncthreads();
  gated_mlp_core<384, 128>(sA, sH, sG, w_inT, normw, wave, lane);
  v8f m = tile_matmul_out128<128, 128>(sG, w_outT, wave, lane);
  int col = wave * 16 + (lane & 15);
  int rofs = (lane >> 4) << 3;
#pragma unroll
  for (int r = 0; r < 8; r++) {
    int e = e0 + rofs + r;
    float u = m[r];
    upd_bf[(size_t)e * 128 + col] = f2bf(u);
    atomicAdd(&node_delta[(size_t)n2e[e] * 128 + col],
              u * env[(size_t)e * 128 + col] * sw[e]);
  }
}

// ---------------------------------------------------------------- dense GEMM (K=128): dst = base + (A@W^T)*scale
template <int OUT>
__global__ __launch_bounds__(256) void k_dense_gemm_add(
    const unsigned short* __restrict__ A, const unsigned short* __restrict__ WT,
    const float* __restrict__ base, float* __restrict__ dst, float scale, int rows) {
  constexpr int NT = OUT / 16;  // 8 or 4
  constexpr int MT = 8 / NT;    // 1 or 2
  int wave = threadIdx.x >> 5, lane = threadIdx.x & 31;
  int mt = wave / NT, nt = wave % NT;
  int row0 = blockIdx.x * (MT * 16) + mt * 16;
  if (row0 >= rows) return;
  const unsigned short* Ab = A + (size_t)row0 * 128;
  v8f acc = {0.f, 0.f, 0.f, 0.f, 0.f, 0.f, 0.f, 0.f};
#pragma unroll
  for (int ks = 0; ks < 4; ks++) {
    v16bf a = load_tileA(Ab, 128, ks, lane);
    v16bf b = load_tileB16(WT, 128, nt * 16, ks, lane);
    acc = wmma_bf16(a, b, acc);
  }
  int col = nt * 16 + (lane & 15);
  int rbase = row0 + ((lane >> 4) << 3);
#pragma unroll
  for (int r = 0; r < 8; r++) {
    size_t o = (size_t)(rbase + r) * OUT + col;
    dst[o] = base[o] + acc[r] * scale;
  }
}

// ---------------------------------------------------------------- streaming / elementwise kernels
template <int D>
__global__ __launch_bounds__(256) void k_rmsnorm_bf16(const float* __restrict__ x,
                                                      const float* __restrict__ w,
                                                      unsigned short* __restrict__ out, int rows) {
  int wave = threadIdx.x >> 5, lane = threadIdx.x & 31;
  int row = blockIdx.x * 8 + wave;
  if (row >= rows) return;
  constexpr int PER = D / 32;
  float v[PER]; float ss = 0.f;
#pragma unroll
  for (int j = 0; j < PER; j++) { v[j] = x[(size_t)row * D + lane + j * 32]; ss += v[j] * v[j]; }
#pragma unroll
  for (int off = 16; off; off >>= 1) ss += __shfl_xor(ss, off, 32);
  float inv = rsqrtf(ss / (float)D + 1e-6f);
#pragma unroll
  for (int j = 0; j < PER; j++)
    out[(size_t)row * D + lane + j * 32] = f2bf(v[j] * inv * w[lane + j * 32]);
}

__global__ void k_f32_to_bf16(const float* __restrict__ x, unsigned short* __restrict__ y, int n) {
  int i = blockIdx.x * 256 + threadIdx.x;
  if (i < n) y[i] = f2bf(x[i]);
}

__global__ void k_fill_f32(float* p, float v, int n) {
  int i = blockIdx.x * 256 + threadIdx.x;
  if (i < n) p[i] = v;
}

__global__ void k_extract_node(const float* __restrict__ ext, float* __restrict__ node, int n) {
  int i = blockIdx.x * 256 + threadIdx.x;
  if (i >= n) return;
  int d = i & 127, r = i >> 7;
  int f = r >> 12, l = r & 4095;
  node[i] = ext[((size_t)(f * kNALL + l)) * 128 + d];
}

__global__ void k_transpose_bf16(const float* __restrict__ w, unsigned short* __restrict__ wt,
                                 int K, int N) {
  int i = blockIdx.x * 256 + threadIdx.x;
  if (i >= K * N) return;
  int n = i % N, k = i / N;
  wt[(size_t)n * K + k] = f2bf(w[i]);
}

template <bool SIG>
__global__ void k_env(const float* __restrict__ rbf, const float* __restrict__ w /*12x128*/,
                      float* __restrict__ out, int nrows) {
  int i = blockIdx.x * 256 + threadIdx.x;
  if (i >= nrows * 128) return;
  int c = i & 127, e = i >> 7;
  float s = 0.f;
#pragma unroll
  for (int k = 0; k < 12; k++) s += rbf[e * 12 + k] * w[k * 128 + c];
  out[i] = SIG ? sigm(s) : s;
}

__global__ void k_exp_sum(float* __restrict__ f, const int* __restrict__ idx,
                          const float* __restrict__ m, float* __restrict__ s, int rows) {
  int i = blockIdx.x * 256 + threadIdx.x;
  if (i >= rows * 128) return;
  int r = i >> 7, d = i & 127;
  float e = expf(f[i] - m[(size_t)idx[r] * 128 + d]);
  f[i] = e;
  atomicAdd(&s[(size_t)idx[r] * 128 + d], e);
}

__global__ void k_apply_seg(float* __restrict__ dst, const float* __restrict__ e,
                            const float* __restrict__ ssum, const unsigned short* __restrict__ msg,
                            const float* __restrict__ wrow, const int* __restrict__ idx, int rows) {
  int i = blockIdx.x * 256 + threadIdx.x;
  if (i >= rows * 128) return;
  int r = i >> 7, d = i & 127;
  int t = idx[r];
  float val = e[i] / (ssum[(size_t)t * 128 + d] + 1e-12f) * bf2f(msg[i]) * wrow[r];
  atomicAdd(&dst[(size_t)t * 128 + d], val);
}

// ---------------------------------------------------------------- host launcher
extern "C" void kernel_launch(void* const* d_in, const int* in_sizes, int n_in,
                              void* d_out, int out_size, void* d_ws, size_t ws_size,
                              hipStream_t stream) {
  (void)in_sizes; (void)n_in; (void)out_size;
  if (ws_size < WS_NEED) return;

  const float* node_ext  = (const float*)d_in[0];
  const float* edge_in   = (const float*)d_in[1];
  const float* angle_in  = (const float*)d_in[2];
  const float* sw        = (const float*)d_in[4];
  const float* a_sw      = (const float*)d_in[5];
  const int*   edge_idx  = (const int*)d_in[6];
  const int*   angle_idx = (const int*)d_in[7];
  const float* edge_rbf  = (const float*)d_in[8];
  const int* n2e = edge_idx;            const int* n_ext2e = edge_idx + kNEDGE;
  const int* n2a = angle_idx;           const int* eij = angle_idx + kNANGLE;
  const int* eik = angle_idx + 2 * kNANGLE;

  // params (insertion order of _make_params)
  const float* p_la_ne   = (const float*)d_in[11];
  const float* p_la_na   = (const float*)d_in[12];
  const float* p_la_win  = (const float*)d_in[13];
  const float* p_la_norm = (const float*)d_in[14];
  const float* p_la_wout = (const float*)d_in[15];
  const float* p_la_gate = (const float*)d_in[16];
  const float* p_aa_nn   = (const float*)d_in[17];
  const float* p_aa_ne   = (const float*)d_in[18];
  const float* p_aa_win  = (const float*)d_in[19];
  const float* p_aa_norm = (const float*)d_in[20];
  const float* p_aa_wout = (const float*)d_in[21];
  const float* p_aa_srcg = (const float*)d_in[22];
  const float* p_ae_win  = (const float*)d_in[23];
  const float* p_ae_norm = (const float*)d_in[24];
  const float* p_ae_wout = (const float*)d_in[25];
  const float* p_lr_ne   = (const float*)d_in[26];
  const float* p_lr_na   = (const float*)d_in[27];
  const float* p_lr_win  = (const float*)d_in[28];
  const float* p_lr_norm = (const float*)d_in[29];
  const float* p_lr_wout = (const float*)d_in[30];
  const float* p_lr_env  = (const float*)d_in[31];
  const float* p_lr_eprj = (const float*)d_in[32];
  const float* p_lr_aprj = (const float*)d_in[33];
  const float* p_ar_nn   = (const float*)d_in[34];
  const float* p_ar_ne   = (const float*)d_in[35];
  const float* p_ar_win  = (const float*)d_in[36];
  const float* p_ar_norm = (const float*)d_in[37];
  const float* p_ar_wout = (const float*)d_in[38];
  const float* p_ar_env  = (const float*)d_in[39];
  const float* p_ar_nprj = (const float*)d_in[40];
  const float* p_ar_eprj = (const float*)d_in[41];

  float* out_node  = (float*)d_out;                         // 8192*128
  float* out_edge  = out_node + (size_t)kNNODE * 128;       // NEDGE*128
  float* out_angle = out_edge + (size_t)kNEDGE * 128;       // NANGLE*64

  char* ws = (char*)d_ws;
  auto U = [&](size_t o) { return (unsigned short*)(ws + o); };
  auto F = [&](size_t o) { return (float*)(ws + o); };

  // ---- transpose+convert all weights to bf16 W^T
  auto T = [&](const float* w, size_t off, int K, int N) {
    int n = K * N;
    k_transpose_bf16<<<(n + 255) / 256, 256, 0, stream>>>(w, U(off), K, N);
  };
  T(p_la_win, WT_LA_IN, 448, 512);   T(p_la_wout, WT_LA_OUT, 256, 128);
  T(p_la_gate, WT_LA_GATE, 448, 128);
  T(p_aa_win, WT_AA_IN, 384, 256);   T(p_aa_wout, WT_AA_OUT, 128, 128);
  T(p_aa_srcg, WT_AA_SRCG, 128, 128);
  T(p_ae_win, WT_AE_IN, 384, 512);   T(p_ae_wout, WT_AE_OUT, 256, 128);
  T(p_lr_win, WT_LR_IN, 448, 512);   T(p_lr_wout, WT_LR_OUT, 256, 128);
  T(p_lr_eprj, WT_LR_EPROJ, 128, 128); T(p_lr_aprj, WT_LR_APROJ, 128, 64);
  T(p_ar_win, WT_AR_IN, 384, 256);   T(p_ar_wout, WT_AR_OUT, 128, 128);
  T(p_ar_nprj, WT_AR_NPROJ, 128, 128); T(p_ar_eprj, WT_AR_EPROJ, 128, 128);

  // ---- init working buffers
  hipMemcpyAsync(out_edge, edge_in, (size_t)kNEDGE * 128 * 4, hipMemcpyDeviceToDevice, stream);
  k_extract_node<<<(kNNODE * 128 + 255) / 256, 256, 0, stream>>>(node_ext, out_node, kNNODE * 128);
  k_f32_to_bf16<<<(kNEXT * 128 + 255) / 256, 256, 0, stream>>>(node_ext, U(O_NODEEXT_BF), kNEXT * 128);

  const int GE = kNEDGE * 128, GA = kNANGLE * 128, GN = kNNODE * 128;

  // ================= Stage 1: line attention =================
  k_rmsnorm_bf16<128><<<kNEDGE / 8, 256, 0, stream>>>(out_edge, p_la_ne, U(O_EDGE_NRM), kNEDGE);
  k_rmsnorm_bf16<64><<<kNANGLE / 8, 256, 0, stream>>>(angle_in, p_la_na, U(O_ANGLE_NRM), kNANGLE);
  k_f32_to_bf16<<<(GN + 255) / 256, 256, 0, stream>>>(out_node, U(O_NODE_BF), GN);
  k_fill_f32<<<(GE + 255) / 256, 256, 0, stream>>>(F(O_SEGM), -3.402823466e38f, GE);
  hipMemsetAsync(F(O_SEGS), 0, (size_t)GE * 4, stream);
  k_line_attn<<<kNANGLE / 16, 256, 0, stream>>>(
      U(O_ANGLE_NRM), U(O_NODE_BF), U(O_EDGE_NRM), n2a, eij, eik,
      U(WT_LA_IN), p_la_norm, U(WT_LA_OUT), U(WT_LA_GATE),
      U(O_MSG_BF), F(O_F), F(O_SEGM));
  k_exp_sum<<<(GA + 255) / 256, 256, 0, stream>>>(F(O_F), eij, F(O_SEGM), F(O_SEGS), kNANGLE);
  k_apply_seg<<<(GA + 255) / 256, 256, 0, stream>>>(out_edge, F(O_F), F(O_SEGS),
                                                    U(O_MSG_BF), a_sw, eij, kNANGLE);

  // ================= Stage 2: atom attention =================
  k_rmsnorm_bf16<128><<<kNNODE / 8, 256, 0, stream>>>(out_node, p_aa_nn, U(O_NODE_BF), kNNODE);
  k_rmsnorm_bf16<128><<<kNEDGE / 8, 256, 0, stream>>>(out_edge, p_aa_ne, U(O_EDGE_NRM), kNEDGE);
  k_fill_f32<<<(GN + 255) / 256, 256, 0, stream>>>(F(O_SEGM), -3.402823466e38f, GN);
  hipMemsetAsync(F(O_SEGS), 0, (size_t)GN * 4, stream);
  k_atom_attn<<<kNEDGE / 16, 256, 0, stream>>>(
      U(O_EDGE_NRM), U(O_NODE_BF), U(O_NODEEXT_BF), n2e, n_ext2e,
      U(WT_AA_IN), p_aa_norm, U(WT_AA_OUT), U(WT_AA_SRCG),
      U(WT_AE_IN), p_ae_norm, U(WT_AE_OUT),
      U(O_EUPD_BF), F(O_F), F(O_SEGM), sw, out_edge);
  k_exp_sum<<<(GE + 255) / 256, 256, 0, stream>>>(F(O_F), n2e, F(O_SEGM), F(O_SEGS), kNEDGE);
  k_apply_seg<<<(GE + 255) / 256, 256, 0, stream>>>(out_node, F(O_F), F(O_SEGS),
                                                    U(O_EUPD_BF), sw, n2e, kNEDGE);

  // ================= Stage 3: line refinement =================
  k_rmsnorm_bf16<128><<<kNEDGE / 8, 256, 0, stream>>>(out_edge, p_lr_ne, U(O_EDGE_NRM), kNEDGE);
  k_rmsnorm_bf16<64><<<kNANGLE / 8, 256, 0, stream>>>(angle_in, p_lr_na, U(O_ANGLE_NRM), kNANGLE);
  k_f32_to_bf16<<<(GN + 255) / 256, 256, 0, stream>>>(out_node, U(O_NODE_BF), GN);
  k_env<false><<<(GE + 255) / 256, 256, 0, stream>>>(edge_rbf, p_lr_env, F(O_F), kNEDGE);
  hipMemsetAsync(F(O_SEGM), 0, (size_t)GE * 4, stream);
  k_line_ref<<<kNANGLE / 16, 256, 0, stream>>>(
      U(O_ANGLE_NRM), U(O_NODE_BF), U(O_EDGE_NRM), n2a, eij, eik,
      U(WT_LR_IN), p_lr_norm, U(WT_LR_OUT),
      F(O_F), a_sw, U(O_MSG_BF), F(O_SEGM));
  k_f32_to_bf16<<<(GE + 255) / 256, 256, 0, stream>>>(F(O_SEGM), U(O_EUPD_BF), GE);
  k_dense_gemm_add<128><<<kNEDGE / 16, 256, 0, stream>>>(
      U(O_EUPD_BF), U(WT_LR_EPROJ), out_edge, out_edge, kASCALE, kNEDGE);
  k_dense_gemm_add<64><<<kNANGLE / 32, 256, 0, stream>>>(
      U(O_MSG_BF), U(WT_LR_APROJ), angle_in, out_angle, 1.0f, kNANGLE);

  // ================= Stage 4: atom refinement =================
  k_rmsnorm_bf16<128><<<kNNODE / 8, 256, 0, stream>>>(out_node, p_ar_nn, U(O_NODE_BF), kNNODE);
  k_rmsnorm_bf16<128><<<kNEDGE / 8, 256, 0, stream>>>(out_edge, p_ar_ne, U(O_EDGE_NRM), kNEDGE);
  k_env<true><<<(GE + 255) / 256, 256, 0, stream>>>(edge_rbf, p_ar_env, F(O_F), kNEDGE);
  hipMemsetAsync(F(O_SEGM), 0, (size_t)GN * 4, stream);
  k_atom_ref<<<kNEDGE / 16, 256, 0, stream>>>(
      U(O_EDGE_NRM), U(O_NODE_BF), U(O_NODEEXT_BF), n2e, n_ext2e,
      U(WT_AR_IN), p_ar_norm, U(WT_AR_OUT),
      sw, F(O_F), U(O_EUPD_BF), F(O_SEGM));
  k_f32_to_bf16<<<(GN + 255) / 256, 256, 0, stream>>>(F(O_SEGM), U(O_NODE_BF), GN);
  k_dense_gemm_add<128><<<kNNODE / 16, 256, 0, stream>>>(
      U(O_NODE_BF), U(WT_AR_NPROJ), out_node, out_node, kESCALE, kNNODE);
  k_dense_gemm_add<128><<<kNEDGE / 16, 256, 0, stream>>>(
      U(O_EUPD_BF), U(WT_AR_EPROJ), out_edge, out_edge, 1.0f, kNEDGE);
}
